// SGCN_5514738008876
// MI455X (gfx1250) — matrix operand order repaired and checked by reading, not compiled, not run on hardware
//
#include <hip/hip_runtime.h>
#include <hip/hip_bf16.h>

// SGCN on MI455X (gfx1250):
//  BN(stat+apply->bf16) -> WMMA bf16 GEMM (async-LDS-staged B, LDS-transposed
//  epilogue) + tanh -> 4x SpMM (bf16 features resident in 192MB L2, packed
//  bf16 device-scope atomics) -> WMMA GEMM+tanh -> WMMA GEMM (f32 logits).

typedef __attribute__((ext_vector_type(16))) __bf16 v16bf;
typedef __attribute__((ext_vector_type(8)))  float  v8f;

#define NNODES 100000
#define DH     256
#define DOUT   64

static __device__ __forceinline__ unsigned short f2bf(float f) {
  unsigned u = __float_as_uint(f);
  u += 0x7FFFu + ((u >> 16) & 1u);        // round-to-nearest-even
  return (unsigned short)(u >> 16);
}
static __device__ __forceinline__ float bf2f(unsigned short h) {
  return __uint_as_float(((unsigned)h) << 16);
}

// ---------------- utility: zero buffers ----------------
__global__ void k_zero_f32(float* __restrict__ p, int n) {
  int i = blockIdx.x * blockDim.x + threadIdx.x;
  if (i < n) p[i] = 0.0f;
}
__global__ void k_zero_u4(uint4* __restrict__ p, long n4) {
  long i = blockIdx.x * (long)blockDim.x + threadIdx.x;
  long s = (long)gridDim.x * blockDim.x;
  uint4 z = make_uint4(0u, 0u, 0u, 0u);
  for (; i < n4; i += s) p[i] = z;
}

// ---------------- BatchNorm statistics ----------------
__global__ void k_bn_partial(const float* __restrict__ x,
                             float* __restrict__ sum, float* __restrict__ sumsq) {
  const int c  = threadIdx.x;
  const int r0 = blockIdx.x * 250;
  float s = 0.f, ss = 0.f;
  for (int r = r0; r < r0 + 250; ++r) {
    float v = x[(long)r * DH + c];
    s += v; ss += v * v;
  }
  atomicAdd(&sum[c], s);
  atomicAdd(&sumsq[c], ss);
}

__global__ void k_bn_finalize(const float* __restrict__ sum, const float* __restrict__ sumsq,
                              const float* __restrict__ gamma, const float* __restrict__ beta,
                              float* __restrict__ scale, float* __restrict__ shift) {
  const int c = threadIdx.x;
  float mean = sum[c] * (1.0f / NNODES);
  float var  = sumsq[c] * (1.0f / NNODES) - mean * mean;
  float rstd = rsqrtf(var + 1e-5f);
  float sc = rstd * gamma[c];
  scale[c] = sc;
  shift[c] = beta[c] - mean * sc;
}

// normalize + convert to bf16 (8 elems / thread)
__global__ void k_bn_apply(const float* __restrict__ x,
                           const float* __restrict__ scale, const float* __restrict__ shift,
                           unsigned int* __restrict__ xb) {
  long t = blockIdx.x * (long)blockDim.x + threadIdx.x;
  long base = t * 8;
  if (base >= (long)NNODES * DH) return;
  int c0 = (int)(base & (DH - 1));
  const float4* src = (const float4*)(x + base);
  float4 a = src[0], b = src[1];
  float f[8] = {a.x, a.y, a.z, a.w, b.x, b.y, b.z, b.w};
  unsigned r[4];
#pragma unroll
  for (int i = 0; i < 4; ++i) {
    float lo = f[2 * i]     * scale[c0 + 2 * i]     + shift[c0 + 2 * i];
    float hi = f[2 * i + 1] * scale[c0 + 2 * i + 1] + shift[c0 + 2 * i + 1];
    r[i] = (unsigned)f2bf(lo) | ((unsigned)f2bf(hi) << 16);
  }
  ((uint4*)xb)[t] = make_uint4(r[0], r[1], r[2], r[3]);
}

// ---------------- pack weights into wave32 WMMA B-fragment layout ----------
// B (32x16 bf16 per K-tile): lanes 0-15 hold column n = nt*16+lane, K = kt*32+0..15
// (dword d = K pair {2d,2d+1}); lanes 16-31 hold same columns, K = kt*32+16..31.
// Linear: wp[(((kt*(Nw/16)+nt)*32 + lane)*8 + d)]
__global__ void k_pack_w(const float* __restrict__ w, unsigned int* __restrict__ wp, int Nw) {
  int idx = blockIdx.x * blockDim.x + threadIdx.x;
  int total = (DH / 32) * (Nw / 16) * 32 * 8;
  if (idx >= total) return;
  int d    = idx & 7;
  int lane = (idx >> 3) & 31;
  int nt   = (idx >> 8) % (Nw / 16);
  int kt   = (idx >> 8) / (Nw / 16);
  int n = nt * 16 + (lane & 15);
  int k = kt * 32 + ((lane & 16) ? 16 : 0) + d * 2;
  unsigned short lo = f2bf(w[(long)k * Nw + n]);
  unsigned short hi = f2bf(w[(long)(k + 1) * Nw + n]);
  wp[idx] = (unsigned)lo | ((unsigned)hi << 16);
}

// ---------------- WMMA GEMM: [NNODES x 256](bf16) @ [256 x NW] + bias -------
// Block = 8 waves x one 64-col group: B panel (32KB) async-staged to LDS once,
// each wave computes a 16x64 tile (4 accumulators, A reused), K-loop of 8.
// bf16 epilogue transposes through a per-wave LDS slice for b128 stores.
template <int NW, bool TANH, bool BF16OUT>
__global__ void k_gemm(const unsigned int* __restrict__ A,   // bf16 pairs, ld=128 dwords
                       const unsigned int* __restrict__ Bp,  // packed B fragments
                       const float* __restrict__ bias,
                       void* __restrict__ Out) {
  __shared__ __align__(16) unsigned int   sB[8192];                      // 32KB B panel
  __shared__ __align__(16) unsigned short sT[BF16OUT ? 8 * 1024 : 8];    // 16KB epilogue

  const int lane = threadIdx.x & 31;
  const int wid  = threadIdx.x >> 5;
  const int NT   = NW / 16;            // n-tiles in full matrix
  const int nb0  = blockIdx.y * 64;    // this block's column base
  const int nt0  = nb0 >> 4;

  // ---- stage this block's B panel into LDS with async copies ----
  {
    const uint4* gB = (const uint4*)Bp;
#pragma unroll
    for (int s = 0; s < 8; ++s) {
      int i   = threadIdx.x + s * 256;          // LDS uint4 index 0..2047
      int kt  = i >> 8;                         // 256 uint4 per K-tile chunk
      int rem = i & 255;
      const uint4* gp = gB + ((kt * NT + nt0) * 64 + rem);
      unsigned ldsoff = (unsigned)(size_t)&sB[i * 4];
      asm volatile("global_load_async_to_lds_b128 %0, %1, off"
                   :: "v"(ldsoff), "v"(gp) : "memory");
    }
    asm volatile("s_wait_asynccnt 0" ::: "memory");
  }
  __syncthreads();

  const int mt = blockIdx.x * 8 + wid;
  if (mt >= NNODES / 16) return;
  const int m0  = mt * 16;
  const int row = m0 + (lane & 15);
  const int koff = (lane & 16) ? 8 : 0;  // dwords: hi lane-group starts at K=16

  union Frag { uint4 q[2]; v16bf v; };
  v8f acc[4] = {};

  const unsigned int* arow = A + (size_t)row * (DH / 2);
#pragma unroll
  for (int kt = 0; kt < DH / 32; ++kt) {
    Frag af;
    const uint4* ap = (const uint4*)(arow + kt * 16 + koff);
    af.q[0] = ap[0];   // K rel {0..7}   ({16..23} for hi lanes)
    af.q[1] = ap[1];   // K rel {8..15} ({24..31})
#pragma unroll
    for (int j = 0; j < 4; ++j) {
      Frag bf;
      const uint4* bp = (const uint4*)&sB[((kt * 4 + j) * 32 + lane) * 8];
      bf.q[0] = bp[0];
      bf.q[1] = bp[1];
      acc[j] = __builtin_amdgcn_wmma_f32_16x16x32_bf16(
          false, af.v, false, bf.v, (short)0, acc[j], false, false);
    }
  }

  if (BF16OUT) {
    // transpose 16x64 bf16 tile through this wave's private LDS slice
    unsigned short* myT = &sT[wid * 1024];
#pragma unroll
    for (int j = 0; j < 4; ++j) {
      float bv = bias[nb0 + j * 16 + (lane & 15)];
#pragma unroll
      for (int v = 0; v < 8; ++v) {
        int rl = (lane & 16) ? (v + 8) : v;
        float val = acc[j][v] + bv;
        if (TANH) val = tanhf(val);
        myT[rl * 64 + j * 16 + (lane & 15)] = f2bf(val);
      }
    }
    asm volatile("s_wait_dscnt 0" ::: "memory");   // intra-wave LDS ordering
    int rl = lane >> 1, half = lane & 1;           // 16 rows x 2 halves
    const uint4* srcT = (const uint4*)&myT[rl * 64 + half * 32];
    uint4* outp = (uint4*)((unsigned short*)Out + (size_t)(m0 + rl) * NW + nb0 + half * 32);
#pragma unroll
    for (int t = 0; t < 4; ++t) outp[t] = srcT[t];
  } else {
#pragma unroll
    for (int j = 0; j < 4; ++j) {
      int col = nb0 + j * 16 + (lane & 15);
      float bv = bias[col];
#pragma unroll
      for (int v = 0; v < 8; ++v) {
        int r = m0 + ((lane & 16) ? (v + 8) : v);
        float val = acc[j][v] + bv;
        if (TANH) val = tanhf(val);
        ((float*)Out)[(size_t)r * NW + col] = val;
      }
    }
  }
}

// ---------------- SpMM hop: y[r] += v * x[c], bf16 features -----------------
// Wave processes 32-edge batches: one coalesced NT load per edge array, then
// v_readlane broadcasts. Per edge each lane loads 8 bf16 (b128), scales, and
// scatters with 4 packed-bf16 device-scope atomics. Features stay L2-resident.
__global__ void k_spmm(const unsigned int* __restrict__ xin,
                       const int* __restrict__ erow, const int* __restrict__ ecol,
                       const float* __restrict__ eval,
                       unsigned int* __restrict__ yout, int E) {
  const int lane = threadIdx.x & 31;
  const int wave = (blockIdx.x * blockDim.x + threadIdx.x) >> 5;
  const int nwaves = (gridDim.x * blockDim.x) >> 5;
  for (long base = (long)wave * 32; base < E; base += (long)nwaves * 32) {
    int cnt = (int)(E - base); if (cnt > 32) cnt = 32;
    int rr = 0, cc = 0; unsigned vv = 0u;
    if (lane < cnt) {
      rr = __builtin_nontemporal_load(&erow[base + lane]);
      cc = __builtin_nontemporal_load(&ecol[base + lane]);
      vv = __float_as_uint(__builtin_nontemporal_load(&eval[base + lane]));
    }
    for (int i = 0; i < cnt; ++i) {
      int   r = __builtin_amdgcn_readlane(rr, i);
      int   c = __builtin_amdgcn_readlane(cc, i);
      float v = __uint_as_float(__builtin_amdgcn_readlane(vv, i));

      const uint4* src = (const uint4*)(xin + (size_t)c * (DH / 2)) + lane;
      uint4 dvec = *src;
      unsigned din[4] = {dvec.x, dvec.y, dvec.z, dvec.w};
      unsigned q[4];
#pragma unroll
      for (int k = 0; k < 4; ++k) {
        float lo = bf2f((unsigned short)(din[k] & 0xFFFFu)) * v;
        float hi = bf2f((unsigned short)(din[k] >> 16)) * v;
        q[k] = (unsigned)f2bf(lo) | ((unsigned)f2bf(hi) << 16);
      }
      unsigned int* dst = yout + (size_t)r * (DH / 2) + lane * 4;
#pragma unroll
      for (int k = 0; k < 4; ++k) {
        asm volatile("global_atomic_pk_add_bf16 %0, %1, off scope:SCOPE_DEV"
                     :: "v"(dst + k), "v"(q[k]) : "memory");
      }
    }
  }
}

// ---------------- host-side orchestration ----------------
extern "C" void kernel_launch(void* const* d_in, const int* in_sizes, int n_in,
                              void* d_out, int out_size, void* d_ws, size_t ws_size,
                              hipStream_t stream) {
  const float* x      = (const float*)d_in[0];
  const int*   erow   = (const int*)d_in[1];
  const int*   ecol   = (const int*)d_in[2];
  const float* eval   = (const float*)d_in[3];
  const float* gamma  = (const float*)d_in[4];
  const float* beta   = (const float*)d_in[5];
  const float* w_in   = (const float*)d_in[6];
  const float* b_in   = (const float*)d_in[7];
  const float* w_conv = (const float*)d_in[8];
  const float* b_conv = (const float*)d_in[9];
  const float* w_out  = (const float*)d_in[10];
  const float* b_out  = (const float*)d_in[11];
  const int E = in_sizes[1];

  const size_t NB2 = (size_t)NNODES * DH * 2;   // one bf16 feature buffer: 51.2 MB
  char* ws = (char*)d_ws;
  unsigned int* xb = (unsigned int*)(ws);             // bf16 normalized x; reused as h2
  unsigned int* h1 = (unsigned int*)(ws + NB2);       // hidden / spmm ping
  unsigned int* t0 = (unsigned int*)(ws + 2 * NB2);   // spmm pong
  unsigned int* wp_in   = (unsigned int*)(ws + 3 * NB2);
  unsigned int* wp_conv = wp_in + 131072 / 4;
  unsigned int* wp_out  = wp_conv + 131072 / 4;
  float* sum   = (float*)(wp_out + 32768 / 4);
  float* sumsq = sum + DH;
  float* scale = sumsq + DH;
  float* shift = scale + DH;

  const long n4 = (long)NNODES * DH / 8;   // uint4 count of one bf16 buffer
  const int  MB = (NNODES / 16 + 7) / 8;   // m-groups of 8 waves -> 782 blocks

  // BatchNorm
  k_zero_f32<<<1, 512, 0, stream>>>(sum, 512);                 // sum + sumsq
  k_bn_partial<<<400, 256, 0, stream>>>(x, sum, sumsq);
  k_bn_finalize<<<1, 256, 0, stream>>>(sum, sumsq, gamma, beta, scale, shift);
  k_bn_apply<<<(NNODES * DH / 8 + 255) / 256, 256, 0, stream>>>(x, scale, shift, xb);

  // Weight packing (tiny)
  k_pack_w<<<128, 256, 0, stream>>>(w_in, wp_in, DH);
  k_pack_w<<<128, 256, 0, stream>>>(w_conv, wp_conv, DH);
  k_pack_w<<<32, 256, 0, stream>>>(w_out, wp_out, DOUT);

  // in_conv + tanh : xb(bf16) @ w_in -> h1(bf16)
  k_gemm<DH, true, true><<<dim3(MB, DH / 64), 256, 0, stream>>>(xb, wp_in, b_in, (void*)h1);

  // 4 SpMM hops, ping-pong h1 <-> t0 (final result lands back in h1)
  k_zero_u4<<<2048, 256, 0, stream>>>((uint4*)t0, n4);
  k_spmm<<<4096, 256, 0, stream>>>(h1, erow, ecol, eval, t0, E);
  k_zero_u4<<<2048, 256, 0, stream>>>((uint4*)h1, n4);
  k_spmm<<<4096, 256, 0, stream>>>(t0, erow, ecol, eval, h1, E);
  k_zero_u4<<<2048, 256, 0, stream>>>((uint4*)t0, n4);
  k_spmm<<<4096, 256, 0, stream>>>(h1, erow, ecol, eval, t0, E);
  k_zero_u4<<<2048, 256, 0, stream>>>((uint4*)h1, n4);
  k_spmm<<<4096, 256, 0, stream>>>(t0, erow, ecol, eval, h1, E);

  // conv linear + tanh : h1 @ w_conv -> xb (reused as h2, bf16)
  k_gemm<DH, true, true><<<dim3(MB, DH / 64), 256, 0, stream>>>(h1, wp_conv, b_conv, (void*)xb);

  // out_conv logits : h2 @ w_out -> d_out (f32)
  k_gemm<DOUT, false, false><<<dim3(MB, 1), 256, 0, stream>>>(xb, wp_out, b_out, d_out);
}
